// BackBone_37623913513320
// MI455X (gfx1250) — compile-verified, matrix operand order checked by reading, not run on hardware
//
#include <hip/hip_runtime.h>
#include <hip/hip_fp16.h>
#include <math.h>
#include <stdint.h>

typedef _Float16 half_t;
typedef __attribute__((ext_vector_type(16))) _Float16 v16h;
typedef __attribute__((ext_vector_type(8)))  _Float16 v8h;
typedef __attribute__((ext_vector_type(8)))  float    v8f;

#define B_    128
#define T_    4096
#define H_    256
#define STEP_ 5
#define GATE_ 768   // 3*H
#define KC_   768   // conv2 im2col K = 256*3
#define SEGW_ 8     // segments per workgroup (conv kernel)
#define NCOL_ 48    // 8*5 columns padded to 3 tiles of 16

// Per-lane WMMA f16 operand fragment: 16 halves = two contiguous 8-half runs
// at k = kbase+hk and k = kbase+16+hk (hk = 8 for upper half-wave), matching
// the CDNA5 16-bit 16x32 A/B layout with row (M or N) = lane&15.
__device__ __forceinline__ v16h load_frag(const half_t* rowptr, int kbase, int lane) {
  const int hk = (lane >> 4) << 3;
  v8h lo = *(const v8h*)(rowptr + kbase + hk);
  v8h hi = *(const v8h*)(rowptr + kbase + 16 + hk);
  v16h r;
#pragma unroll
  for (int i = 0; i < 8; ++i) { r[i] = lo[i]; r[i + 8] = hi[i]; }
  return r;
}

__device__ __forceinline__ float sigmoidf_(float x) { return 1.0f / (1.0f + expf(-x)); }

// ---------------------------------------------------------------------------
// K0: per-batch segment counts, prefix offsets, rotation params
// ---------------------------------------------------------------------------
__global__ void setup_kernel(const float* __restrict__ traj, const int* __restrict__ len,
                             int* __restrict__ segs, int* __restrict__ segoff,
                             float* __restrict__ rotp) {
  __shared__ int ssegs[B_];
  int b = threadIdx.x;
  if (b < B_) {
    int L = len[b];
    int sg = L / STEP_;
    ssegs[b] = sg;
    segs[b] = sg;
    const float* last = traj + ((size_t)b * T_ + (size_t)(L - 1)) * 3;
    float heading = -last[2];
    float th = heading * 0.017453292519943295f;
    rotp[b * 5 + 0] = cosf(th);
    rotp[b * 5 + 1] = sinf(th);
    rotp[b * 5 + 2] = last[0];
    rotp[b * 5 + 3] = last[1];
    rotp[b * 5 + 4] = heading;
  }
  __syncthreads();
  if (b == 0) {
    int acc = 0;
    for (int i = 0; i < B_; ++i) { segoff[i] = acc; acc += ssegs[i]; }
    segoff[B_] = acc;
  }
}

// ---------------------------------------------------------------------------
// K1: f32 -> f16 weight conversion
// ---------------------------------------------------------------------------
__global__ void cvt_kernel(const float* __restrict__ src, half_t* __restrict__ dst, int n) {
  int i = blockIdx.x * blockDim.x + threadIdx.x;
  if (i < n) dst[i] = (half_t)src[i];
}

// ---------------------------------------------------------------------------
// K2: gather+rotate -> conv1 (VALU) -> im2col -> conv2 as WMMA GEMM -> mean
//     One WG = 8 segments = 40 real columns (padded to 48).
//     GEMM: O[256, 48] = W2[256, 768] @ Xcol, f16 in / f32 acc.
//     Register-tiled: each wave owns 2 M-tiles x 3 N-tiles, K outermost.
// ---------------------------------------------------------------------------
__global__ void conv_kernel(const float* __restrict__ traj,
                            const int* __restrict__ len,
                            const int* __restrict__ segoff,
                            const float* __restrict__ rotp,
                            const float* __restrict__ w1, const float* __restrict__ b1,
                            const half_t* __restrict__ w2h, const float* __restrict__ b2,
                            half_t* __restrict__ ar, int ntot) {
  extern __shared__ char smem[];
  int*    s_off  = (int*)smem;                                   // 129 ints
  float*  s_xin  = (float*)(smem + 528);                         // [8*5][3]
  half_t* s_xcol = (half_t*)(smem + 1968);                       // [48][768] f16
  float*  s_ot   = (float*)(smem + 1968 + NCOL_ * KC_ * 2);      // [48][256] f32

  const int tid  = threadIdx.x;
  const int seg0 = blockIdx.x * SEGW_;

  for (int i = tid; i < B_ + 1; i += blockDim.x) s_off[i] = segoff[i];
  __syncthreads();

  // ---- stage + rotate raw inputs: one thread per (segment, t) ----
  if (tid < SEGW_ * STEP_) {
    int sl = tid / STEP_, t = tid % STEP_;
    int g = seg0 + sl;
    float vx = 0.f, vy = 0.f, va = 0.f;
    if (g < ntot) {
      int lo = 0, hi = B_ - 1;                 // find batch owning segment g
      while (lo < hi) { int mid = (lo + hi + 1) >> 1; if (s_off[mid] <= g) lo = mid; else hi = mid - 1; }
      int b = lo;
      int sidx = g - s_off[b];
      int L = len[b];
      int tt = (L % STEP_) + STEP_ * sidx + t;
      const float* p = traj + ((size_t)b * T_ + (size_t)tt) * 3;
      float c = rotp[b * 5 + 0], sn = rotp[b * 5 + 1];
      float ox = rotp[b * 5 + 2], oy = rotp[b * 5 + 3], hd = rotp[b * 5 + 4];
      float x = p[0] - ox, y = p[1] - oy;
      vx = c * x - sn * y;
      vy = sn * x + c * y;
      float a = fmodf(p[2] + hd + 720.0f, 360.0f) * 0.017453292519943295f;
      if (a > 3.14159265358979f) a -= 6.283185307179586f;
      va = a;
    }
    float* q = s_xin + (sl * STEP_ + t) * 3;
    q[0] = vx; q[1] = vy; q[2] = va;
  }
  __syncthreads();

  // ---- conv1 (3->256, k=3, SAME) + relu, scatter into im2col (thread = out ch) ----
  {
    const int c = tid;  // 0..255
    float w[9];
#pragma unroll
    for (int i = 0; i < 9; ++i) w[i] = w1[c * 9 + i];
    float bb = b1[c];
    for (int sl = 0; sl < SEGW_; ++sl) {
      float xv[STEP_ + 2][3];
#pragma unroll
      for (int ci = 0; ci < 3; ++ci) { xv[0][ci] = 0.f; xv[STEP_ + 1][ci] = 0.f; }
#pragma unroll
      for (int t = 0; t < STEP_; ++t)
#pragma unroll
        for (int ci = 0; ci < 3; ++ci) xv[t + 1][ci] = s_xin[(sl * STEP_ + t) * 3 + ci];
#pragma unroll
      for (int t = 0; t < STEP_; ++t) {
        float acc = bb;
#pragma unroll
        for (int ci = 0; ci < 3; ++ci)
#pragma unroll
          for (int k = 0; k < 3; ++k) acc += w[ci * 3 + k] * xv[t + k][ci];
        acc = fmaxf(acc, 0.f);
        // Xcol[(c,k)][t2] = x1[c][t2+k-1]  ->  scatter this x1 value (t = t2+k-1)
#pragma unroll
        for (int k = 0; k < 3; ++k) {
          int t2 = t + 1 - k;
          if (t2 >= 0 && t2 < STEP_)
            s_xcol[(size_t)(sl * STEP_ + t2) * KC_ + c * 3 + k] = (half_t)acc;
        }
      }
      // entries referencing x1 outside [0,5) are the SAME-pad zeros:
      s_xcol[(size_t)(sl * STEP_ + 0) * KC_ + c * 3 + 0] = (half_t)0.f;
      s_xcol[(size_t)(sl * STEP_ + 4) * KC_ + c * 3 + 2] = (half_t)0.f;
    }
    // zero the 8 padding columns (40..47) across the full K dim
    for (int col = SEGW_ * STEP_; col < NCOL_; ++col) {
#pragma unroll
      for (int k = 0; k < 3; ++k) s_xcol[(size_t)col * KC_ + c * 3 + k] = (half_t)0.f;
    }
  }
  __syncthreads();

  // ---- conv2 GEMM via WMMA, register-tiled 2 (M) x 3 (N) per wave ----
  {
    const int wave = tid >> 5, lane = tid & 31;
    const int row  = lane & 15;
    const int mt0  = wave * 2;                          // 8 waves -> 16 M-tiles
    const half_t* a0 = w2h + (size_t)((mt0 + 0) * 16 + row) * KC_;
    const half_t* a1 = w2h + (size_t)((mt0 + 1) * 16 + row) * KC_;
    v8f acc[2][3];
#pragma unroll
    for (int j = 0; j < 2; ++j)
#pragma unroll
      for (int nt = 0; nt < 3; ++nt)
        acc[j][nt] = (v8f){0.f, 0.f, 0.f, 0.f, 0.f, 0.f, 0.f, 0.f};

    for (int kb = 0; kb < KC_; kb += 32) {
      v16h fa0 = load_frag(a0, kb, lane);
      v16h fa1 = load_frag(a1, kb, lane);
#pragma unroll
      for (int nt = 0; nt < 3; ++nt) {
        v16h fb = load_frag(s_xcol + (size_t)(nt * 16 + row) * KC_, kb, lane);
        acc[0][nt] = __builtin_amdgcn_wmma_f32_16x16x32_f16(false, fa0, false, fb, (short)0, acc[0][nt], false, false);
        acc[1][nt] = __builtin_amdgcn_wmma_f32_16x16x32_f16(false, fa1, false, fb, (short)0, acc[1][nt], false, false);
      }
    }
    const int n = lane & 15, mhi = (lane >> 4) << 3;
#pragma unroll
    for (int j = 0; j < 2; ++j)
#pragma unroll
      for (int nt = 0; nt < 3; ++nt)
#pragma unroll
        for (int r = 0; r < 8; ++r)
          s_ot[(size_t)(nt * 16 + n) * H_ + (mt0 + j) * 16 + mhi + r] = acc[j][nt][r];
  }
  __syncthreads();

  // ---- bias + relu + mean over 5 -> ar[g][c] (f16) ----
  {
    const int c = tid;
    float bb = b2[c];
    for (int sl = 0; sl < SEGW_; ++sl) {
      int g = seg0 + sl;
      if (g >= ntot) break;
      float sum = 0.f;
#pragma unroll
      for (int t = 0; t < STEP_; ++t)
        sum += fmaxf(s_ot[(size_t)(sl * STEP_ + t) * H_ + c] + bb, 0.f);
      ar[(size_t)g * H_ + c] = (half_t)(sum * 0.2f);
    }
  }
}

// ---------------------------------------------------------------------------
// K3: GRU scan. The recurrence is independent per batch row, so 8 WGs each
//     own 16 rows. Per step: GX = x@Wih^T and GH = h@Whh^T (M=16,N=768,K=256)
//     via WMMA with register-tiled accumulators (6 N-tiles per wave per GEMM).
//     x_t is staged into LDS with GLOBAL_LOAD_ASYNC_TO_LDS_B128 (ASYNCcnt),
//     overlapped with the f32->f16 h conversion. Rows past a sample's segment
//     count are masked in the gate stage, so clamped garbage x is harmless.
// ---------------------------------------------------------------------------
__global__ void gru_kernel(const half_t* __restrict__ ar,
                           const int* __restrict__ segs,
                           const int* __restrict__ segoff,
                           const half_t* __restrict__ wih,
                           const half_t* __restrict__ whh,
                           const float* __restrict__ bih,
                           const float* __restrict__ bhh,
                           float* __restrict__ out, int ntot) {
  extern __shared__ char smem[];
  half_t* s_x  = (half_t*)smem;              // [16][256] f16
  half_t* s_h  = s_x + 16 * H_;              // [16][256] f16
  float*  s_hf = (float*)(s_h + 16 * H_);    // [16][256] f32 state
  float*  s_g  = s_hf + 16 * H_;             // [2][16][768] f32 (GX, GH)
  __shared__ int s_segs[16], s_base[16], s_smax;

  const int tid = threadIdx.x;
  const int b0  = blockIdx.x * 16;
  if (tid < 16) { s_segs[tid] = segs[b0 + tid]; s_base[tid] = segoff[b0 + tid]; }
  for (int i = tid; i < 16 * H_; i += blockDim.x) s_hf[i] = 0.f;
  __syncthreads();
  if (tid == 0) {
    int mx = 0;
    for (int i = 0; i < 16; ++i) mx = max(mx, s_segs[i]);
    s_smax = mx;
  }
  __syncthreads();
  const int smax = s_smax;
  const int wave = tid >> 5, lane = tid & 31;

  for (int s = 0; s < smax; ++s) {
    // ---- issue async x_t staging (16 rows x 512B, b128 chunks) ----
#pragma unroll
    for (int it = 0; it < 2; ++it) {
      int i   = tid + it * 256;          // 0..511 chunk id
      int row = i >> 5;                  // 0..15
      int chk = i & 31;                  // 16B chunk within row
      int idx = s_base[row] + s;
      if (idx >= ntot) idx = ntot - 1;   // clamped: masked rows never used
      const half_t* gp = ar + (size_t)idx * H_ + chk * 8;
      uint32_t ldsa = (uint32_t)(uintptr_t)(s_x + row * H_ + chk * 8);
      asm volatile("global_load_async_to_lds_b128 %0, %1, off"
                   :: "v"(ldsa), "v"((uint64_t)(uintptr_t)gp)
                   : "memory");
      // warm L2 for next step while this step's GEMMs run
      int idxn = idx + 1 < ntot ? idx + 1 : idx;
      __builtin_prefetch(ar + (size_t)idxn * H_ + chk * 8, 0, 1);
    }
    // ---- overlap: h f32 -> f16 conversion ----
    {
      int row = tid >> 4;
      int c0  = (tid & 15) << 4;
#pragma unroll
      for (int j = 0; j < 16; ++j)
        s_h[row * H_ + c0 + j] = (half_t)s_hf[row * H_ + c0 + j];
    }
    asm volatile("s_wait_asynccnt 0x0" ::: "memory");
    __syncthreads();

    // ---- two GEMMs, register-tiled: 6 N-tiles per wave per GEMM ----
#pragma unroll
    for (int g = 0; g < 2; ++g) {
      const half_t* arow = (g ? s_h : s_x) + (size_t)(lane & 15) * H_;
      const half_t* wb   = g ? whh : wih;
      v8f acc[6];
#pragma unroll
      for (int i = 0; i < 6; ++i) acc[i] = (v8f){0.f, 0.f, 0.f, 0.f, 0.f, 0.f, 0.f, 0.f};
#pragma unroll
      for (int kb = 0; kb < H_; kb += 32) {
        v16h a = load_frag(arow, kb, lane);
#pragma unroll
        for (int i = 0; i < 6; ++i) {
          int nt = wave * 6 + i;
          v16h b = load_frag(wb + (size_t)(nt * 16 + (lane & 15)) * H_, kb, lane);
          acc[i] = __builtin_amdgcn_wmma_f32_16x16x32_f16(false, a, false, b, (short)0, acc[i], false, false);
        }
      }
      const int n = lane & 15, mhi = (lane >> 4) << 3;
#pragma unroll
      for (int i = 0; i < 6; ++i) {
        int nt = wave * 6 + i;
#pragma unroll
        for (int r = 0; r < 8; ++r)
          s_g[(size_t)g * 16 * GATE_ + (size_t)(mhi + r) * GATE_ + nt * 16 + n] = acc[i][r];
      }
    }
    __syncthreads();

    // ---- gate math: ch = tid, rows 0..15 ----
    {
      const int ch = tid;
      for (int row = 0; row < 16; ++row) {
        float gxr = s_g[row * GATE_ + ch]       + bih[ch];
        float gxz = s_g[row * GATE_ + 256 + ch] + bih[256 + ch];
        float gxn = s_g[row * GATE_ + 512 + ch] + bih[512 + ch];
        float ghr = s_g[16 * GATE_ + row * GATE_ + ch]       + bhh[ch];
        float ghz = s_g[16 * GATE_ + row * GATE_ + 256 + ch] + bhh[256 + ch];
        float ghn = s_g[16 * GATE_ + row * GATE_ + 512 + ch] + bhh[512 + ch];
        float r_ = sigmoidf_(gxr + ghr);
        float z_ = sigmoidf_(gxz + ghz);
        float n_ = tanhf(gxn + r_ * ghn);
        float hp = s_hf[row * H_ + ch];
        float hn = (1.f - z_) * n_ + z_ * hp;
        if (s < s_segs[row]) {
          s_hf[row * H_ + ch] = hn;
          out[((size_t)(s_base[row] + s)) * H_ + ch] = hn;
        }
      }
    }
    __syncthreads();
  }
}

// ---------------------------------------------------------------------------
extern "C" void kernel_launch(void* const* d_in, const int* in_sizes, int n_in,
                              void* d_out, int out_size, void* d_ws, size_t ws_size,
                              hipStream_t stream) {
  const float* traj = (const float*)d_in[0];
  const int*   len  = (const int*)d_in[1];
  const float* w1   = (const float*)d_in[2];
  const float* b1   = (const float*)d_in[3];
  const float* w2   = (const float*)d_in[4];
  const float* b2   = (const float*)d_in[5];
  const float* wihf = (const float*)d_in[6];
  const float* whhf = (const float*)d_in[7];
  const float* bih  = (const float*)d_in[8];
  const float* bhh  = (const float*)d_in[9];
  float* out = (float*)d_out;

  char* ws = (char*)d_ws;
  int*    segs   = (int*)ws;                               // 128 ints
  int*    segoff = (int*)(ws + 512);                       // 129 ints
  float*  rotp   = (float*)(ws + 1536);                    // 128*5 floats
  half_t* w2h    = (half_t*)(ws + 4096);                   // 256*768 f16
  half_t* wihh   = (half_t*)(ws + 4096 + 393216);          // 768*256 f16
  half_t* whhh   = (half_t*)(ws + 4096 + 2 * 393216);      // 768*256 f16
  half_t* ar     = (half_t*)(ws + 4096 + 3 * 393216);      // [ntot][256] f16

  const int ntot = out_size / H_;
  const int NW   = 256 * 768;

  setup_kernel<<<1, 128, 0, stream>>>(traj, len, segs, segoff, rotp);
  cvt_kernel<<<(NW + 255) / 256, 256, 0, stream>>>(w2, w2h, NW);
  cvt_kernel<<<(NW + 255) / 256, 256, 0, stream>>>(wihf, wihh, NW);
  cvt_kernel<<<(NW + 255) / 256, 256, 0, stream>>>(whhf, whhh, NW);

  int nblk = (ntot + SEGW_ - 1) / SEGW_;
  size_t conv_lds = 1968 + (size_t)NCOL_ * KC_ * 2 + (size_t)NCOL_ * H_ * 4;   // ~125 KB
  conv_kernel<<<nblk, 256, conv_lds, stream>>>(traj, len, segoff, rotp,
                                               w1, b1, w2h, b2, ar, ntot);

  size_t gru_lds = (size_t)16 * H_ * 2 * 2 + (size_t)16 * H_ * 4
                 + (size_t)2 * 16 * GATE_ * 4;                                 // ~128 KB
  gru_kernel<<<8, 256, gru_lds, stream>>>(ar, segs, segoff, wihh, whhh, bih, bhh, out, ntot);
}
